// pointnet2_cls_44787918963487
// MI455X (gfx1250) — compile-verified, matrix operand order checked by reading, not run on hardware
//
#include <hip/hip_runtime.h>
#include <hip/hip_bf16.h>

typedef unsigned short u16;
typedef u16 v8u16  __attribute__((ext_vector_type(8)));
typedef u16 v16u16 __attribute__((ext_vector_type(16)));
typedef __bf16 v16bf __attribute__((ext_vector_type(16)));
typedef float v8f   __attribute__((ext_vector_type(8)));

__device__ __forceinline__ u16 f2bf(float f) {
  union { float f; unsigned u; } v; v.f = f;
  unsigned r = v.u + 0x7fffu + ((v.u >> 16) & 1u);
  return (u16)(r >> 16);
}
__device__ __forceinline__ float bf2f(u16 h) {
  union { unsigned u; float f; } v; v.u = ((unsigned)h) << 16; return v.f;
}

// ---------------- xyz (B,3,N) -> pts (B,N,3) ----------------
__global__ void k_transpose(const float* __restrict__ xyz, float* __restrict__ pts, int N) {
  int i = blockIdx.x * blockDim.x + threadIdx.x;
  if (i >= 16 * N) return;
  int b = i / N, n = i % N;
  pts[(size_t)(b * N + n) * 3 + 0] = xyz[(size_t)(b * 3 + 0) * N + n];
  pts[(size_t)(b * N + n) * 3 + 1] = xyz[(size_t)(b * 3 + 1) * N + n];
  pts[(size_t)(b * N + n) * 3 + 2] = xyz[(size_t)(b * 3 + 2) * N + n];
}

// ---------------- farthest point sampling: one block per batch ----------------
__global__ void k_fps(const float* __restrict__ pts, int N, int S, int* __restrict__ out) {
  __shared__ float sdist[4096];
  __shared__ float sval[256];
  __shared__ int   sidx[256];
  __shared__ int   sfar;
  int b = blockIdx.x, t = threadIdx.x;
  for (int i = t; i < N; i += 256) sdist[i] = 1e10f;
  __syncthreads();
  int far = 0;
  for (int j = 0; j < S; ++j) {
    if (t == 0) out[b * S + j] = far;
    const float* cp = pts + ((size_t)b * N + far) * 3;
    float cx = cp[0], cy = cp[1], cz = cp[2];
    float best = -1.0f; int bi = 0;
    for (int i = t; i < N; i += 256) {
      const float* p = pts + ((size_t)b * N + i) * 3;
      float dx = p[0] - cx, dy = p[1] - cy, dz = p[2] - cz;
      float d = dx * dx + dy * dy + dz * dz;
      float nd = fminf(sdist[i], d);
      sdist[i] = nd;
      if (nd > best) { best = nd; bi = i; }
    }
    sval[t] = best; sidx[t] = bi;
    __syncthreads();
    for (int o = 128; o > 0; o >>= 1) {
      if (t < o) {
        if (sval[t + o] > sval[t] ||
            (sval[t + o] == sval[t] && sidx[t + o] < sidx[t])) {
          sval[t] = sval[t + o]; sidx[t] = sidx[t + o];
        }
      }
      __syncthreads();
    }
    if (t == 0) sfar = sidx[0];
    __syncthreads();
    far = sfar;
  }
}

// ---------------- gather sampled centers ----------------
__global__ void k_gather_xyz(const float* __restrict__ pts, const int* __restrict__ fps,
                             float* __restrict__ nxyz, int N, int S) {
  int i = blockIdx.x * blockDim.x + threadIdx.x;
  if (i >= 16 * S) return;
  int b = i / S, s = i % S;
  int n = fps[b * S + s];
  nxyz[(size_t)i * 3 + 0] = pts[((size_t)b * N + n) * 3 + 0];
  nxyz[(size_t)i * 3 + 1] = pts[((size_t)b * N + n) * 3 + 1];
  nxyz[(size_t)i * 3 + 2] = pts[((size_t)b * N + n) * 3 + 2];
}

// ---------------- ball query: first 32 in-radius indices (ascending) ----------------
__global__ void k_ballquery(const float* __restrict__ pts, const float* __restrict__ nxyz,
                            int* __restrict__ idx, int N, int S, float r2) {
  int i = blockIdx.x * blockDim.x + threadIdx.x;
  if (i >= 16 * S) return;
  int b = i / S;
  float cx = nxyz[(size_t)i * 3 + 0];
  float cy = nxyz[(size_t)i * 3 + 1];
  float cz = nxyz[(size_t)i * 3 + 2];
  int cnt = 0, first = 0;
  for (int n = 0; n < N && cnt < 32; ++n) {
    const float* p = pts + ((size_t)b * N + n) * 3;
    float dx = p[0] - cx, dy = p[1] - cy, dz = p[2] - cz;
    float d = dx * dx + dy * dy + dz * dz;
    if (d <= r2) {
      if (cnt == 0) first = n;
      idx[(size_t)i * 32 + cnt] = n;
      ++cnt;
    }
  }
  for (; cnt < 32; ++cnt) idx[(size_t)i * 32 + cnt] = first;
}

// ---------------- build grouped input X (M=B*S*32, Cpad) in bf16 ----------------
__global__ void k_group(const float* __restrict__ pts, const float* __restrict__ nxyz,
                        const u16* __restrict__ feats, const int* __restrict__ idx,
                        u16* __restrict__ X, int N, int S, int Cfeat, int Cpad, long total) {
  long i = (long)blockIdx.x * blockDim.x + threadIdx.x;
  if (i >= total) return;
  int  c   = (int)(i % Cpad);
  long row = i / Cpad;
  int  k   = (int)(row & 31);
  long bs  = row >> 5;
  int  b   = (int)(bs / S);
  if (c < 3) {
    int n = idx[bs * 32 + k];
    X[i] = f2bf(pts[((size_t)b * N + n) * 3 + c] - nxyz[bs * 3 + c]);
  } else if (c < 3 + Cfeat) {
    int n = idx[bs * 32 + k];
    X[i] = feats[((size_t)b * N + n) * Cfeat + (c - 3)];
  } else {
    X[i] = 0;
  }
}

// ---------------- WMMA GEMM: Y(M,N) = X(M,Kp) * W(N,Kp)^T + bias, bf16 in, bf16 out ----------------
__global__ void k_gemm_bf16(const u16* __restrict__ X, const u16* __restrict__ W,
                            const float* __restrict__ bias, u16* __restrict__ Y,
                            int M, int Kp, int N) {
  int wave = threadIdx.x >> 5;
  int lane = threadIdx.x & 31;
  int tilesM = M >> 4;
  int tm = blockIdx.x * 4 + wave;
  int tn = blockIdx.y;
  if (tm >= tilesM) return;            // uniform per wave: EXEC stays all-ones for WMMA
  int h = lane >> 4;
  int mrow = (tm << 4) + (lane & 15);
  int nrow = (tn << 4) + (lane & 15);
  const u16* xp = X + (size_t)mrow * Kp;
  const u16* wp = W + (size_t)nrow * Kp;
  v8f acc = {};
  for (int k0 = 0; k0 < Kp; k0 += 32) {
    // 16-bit A (16x32) fragment layout: lane holds row M=lane%16,
    // VGPR0..3 = K in [h*8, h*8+8), VGPR4..7 = K in [16+h*8, 16+h*8+8)
    v8u16 alo = *(const v8u16*)(xp + k0 + h * 8);
    v8u16 ahi = *(const v8u16*)(xp + k0 + 16 + h * 8);
    v8u16 blo = *(const v8u16*)(wp + k0 + h * 8);
    v8u16 bhi = *(const v8u16*)(wp + k0 + 16 + h * 8);
    v16u16 au, bu;
#pragma unroll
    for (int q = 0; q < 8; ++q) {
      au[q] = alo[q]; au[q + 8] = ahi[q];
      bu[q] = blo[q]; bu[q + 8] = bhi[q];
    }
    acc = __builtin_amdgcn_wmma_f32_16x16x32_bf16(
        false, __builtin_bit_cast(v16bf, au),
        false, __builtin_bit_cast(v16bf, bu),
        (short)0, acc, false, false);
  }
  float bv = bias[nrow];
#pragma unroll
  for (int j = 0; j < 8; ++j) {
    int m = (tm << 4) + j + 8 * h;     // C/D layout: VGPR j -> M=j (lanes 0-15), M=8+j (lanes 16-31)
    Y[(size_t)m * N + nrow] = f2bf(acc[j] + bv);
  }
}

// ---------------- BN statistics ----------------
__global__ void k_zero_stats(float* __restrict__ stats, int C) {
  int i = blockIdx.x * blockDim.x + threadIdx.x;
  if (i < 2 * C) stats[i] = 0.0f;
}

__global__ void k_stats_acc(const u16* __restrict__ Y, float* __restrict__ stats, int M, int C) {
  int c = blockIdx.x;
  int t = threadIdx.x;
  long base = (long)blockIdx.y * 256 * 16;
  float s = 0.0f, s2 = 0.0f;
  for (int q = 0; q < 16; ++q) {
    long r = base + t + (long)q * 256;
    if (r < M) {
      float v = bf2f(Y[r * C + c]);
      s += v; s2 += v * v;
    }
  }
  __shared__ float sh[256], sh2[256];
  sh[t] = s; sh2[t] = s2;
  __syncthreads();
  for (int o = 128; o > 0; o >>= 1) {
    if (t < o) { sh[t] += sh[t + o]; sh2[t] += sh2[t + o]; }
    __syncthreads();
  }
  if (t == 0) {
    atomicAdd(&stats[c], sh[0]);
    atomicAdd(&stats[C + c], sh2[0]);
  }
}

// ---------------- BN + ReLU, in place on bf16 buffer ----------------
__global__ void k_bnrelu(u16* __restrict__ Y, const float* __restrict__ stats,
                         const float* __restrict__ gamma, const float* __restrict__ beta,
                         long M, int C) {
  long i = (long)blockIdx.x * blockDim.x + threadIdx.x;
  if (i >= M * C) return;
  int c = (int)(i % C);
  float mean = stats[c] / (float)M;
  float var  = stats[C + c] / (float)M - mean * mean;
  float y = bf2f(Y[i]);
  float o = gamma[c] * (y - mean) * rsqrtf(var + 1e-5f) + beta[c];
  Y[i] = f2bf(fmaxf(o, 0.0f));
}

// ---------------- max over K=32 neighbors ----------------
__global__ void k_maxpool_k(const u16* __restrict__ Y, u16* __restrict__ feats, int SB, int C) {
  long i = (long)blockIdx.x * blockDim.x + threadIdx.x;
  if (i >= (long)SB * C) return;
  int  c  = (int)(i % C);
  long bs = i / C;
  float m = -1e30f;
  for (int k = 0; k < 32; ++k) m = fmaxf(m, bf2f(Y[(bs * 32 + k) * C + c]));
  feats[i] = f2bf(m);
}

// ---------------- max over S centers (head input) ----------------
__global__ void k_maxpool_s(const u16* __restrict__ feats, u16* __restrict__ pooled, int S, int C) {
  int i = blockIdx.x * blockDim.x + threadIdx.x;
  if (i >= 16 * C) return;
  int c = i % C, b = i / C;
  float m = -1e30f;
  for (int s = 0; s < S; ++s) m = fmaxf(m, bf2f(feats[((size_t)b * S + s) * C + c]));
  pooled[i] = f2bf(m);
}

// ---------------- convert + pad weights: f32 (Ct,Kt) -> bf16 (Nout,Kp) ----------------
__global__ void k_convw(const float* __restrict__ W, u16* __restrict__ Wb,
                        int Nout, int Kp, int Ct, int Kt) {
  int i = blockIdx.x * blockDim.x + threadIdx.x;
  if (i >= Nout * Kp) return;
  int n = i / Kp, k = i % Kp;
  Wb[i] = (n < Ct && k < Kt) ? f2bf(W[(size_t)n * Kt + k]) : (u16)0;
}

__global__ void k_padbias(const float* __restrict__ b, float* __restrict__ bp, int Ct, int Np) {
  int i = threadIdx.x;
  if (i < Np) bp[i] = (i < Ct) ? b[i] : 0.0f;
}

// ---------------- final logits: relu + narrow 48 -> 40, write f32 ----------------
__global__ void k_final(const u16* __restrict__ Yh, float* __restrict__ out) {
  int i = blockIdx.x * blockDim.x + threadIdx.x;
  if (i >= 16 * 40) return;
  int b = i / 40, c = i % 40;
  out[i] = fmaxf(bf2f(Yh[b * 48 + c]), 0.0f);
}

extern "C" void kernel_launch(void* const* d_in, const int* in_sizes, int n_in,
                              void* d_out, int out_size, void* d_ws, size_t ws_size,
                              hipStream_t stream) {
  (void)in_sizes; (void)n_in; (void)out_size; (void)ws_size;
  const int B = 16;
  const int   Ns[4]    = {4096, 1024, 256, 64};
  const int   Ss[4]    = {1024, 256, 64, 16};
  const float R2[4]    = {0.01f, 0.04f, 0.16f, 0.64f};
  const int   Cfeat[4] = {0, 64, 128, 256};
  const int   Cin0[4]  = {3, 67, 131, 259};
  const int   Cpad0[4] = {32, 96, 160, 288};
  const int   CH[4][3] = {{32, 32, 64}, {64, 64, 128}, {128, 128, 256}, {256, 256, 512}};

  char* ws = (char*)d_ws;
  size_t off = 0;
  auto alloc = [&](size_t bytes) -> size_t {
    size_t o = off; off = (off + bytes + 255) & ~(size_t)255; return o;
  };
  size_t o_pts0 = alloc((size_t)B * 4096 * 3 * 4);
  size_t o_nxyz[4]; for (int s = 0; s < 4; ++s) o_nxyz[s] = alloc((size_t)B * Ss[s] * 3 * 4);
  size_t o_fps = alloc((size_t)B * 1024 * 4);
  size_t o_idx = alloc((size_t)B * 1024 * 32 * 4);
  size_t o_feats[4]; for (int s = 0; s < 4; ++s) o_feats[s] = alloc((size_t)B * Ss[s] * CH[s][2] * 2);
  size_t o_X  = alloc((size_t)524288 * 32 * 2);   // grouped input, max stage0
  size_t o_Y1 = alloc((size_t)524288 * 64 * 2);   // layer1 / layer3 outputs
  size_t o_Y2 = alloc((size_t)524288 * 32 * 2);   // layer2 outputs
  size_t o_stats  = alloc(2 * 512 * 4);
  size_t o_bias48 = alloc(64 * 4);
  size_t o_pool   = alloc(16 * 512 * 2);
  size_t o_hY[3];
  o_hY[0] = alloc(16 * 256 * 2);
  o_hY[1] = alloc(16 * 128 * 2);
  o_hY[2] = alloc(16 * 48 * 2);

  // ----- per-GEMM weight descriptors: 12 SA layers + 3 head layers -----
  int wNp[15], wKp[15], wNt[15], wKt[15];
  const float* wSrc[15];
  const float* bSrc[15];
  for (int s = 0; s < 4; ++s)
    for (int l = 0; l < 3; ++l) {
      int li = s * 3 + l;
      int base = 1 + s * 12 + l * 4;        // d_in: [W, b, gamma, beta] per layer
      wSrc[li] = (const float*)d_in[base];
      bSrc[li] = (const float*)d_in[base + 1];
      wNp[li] = CH[s][l];  wNt[li] = CH[s][l];
      wKt[li] = (l == 0) ? Cin0[s] : CH[s][l - 1];
      wKp[li] = (l == 0) ? Cpad0[s] : CH[s][l - 1];
    }
  wSrc[12] = (const float*)d_in[49]; bSrc[12] = (const float*)d_in[50];
  wNp[12] = 256; wNt[12] = 256; wKp[12] = 512; wKt[12] = 512;
  wSrc[13] = (const float*)d_in[53]; bSrc[13] = (const float*)d_in[54];
  wNp[13] = 128; wNt[13] = 128; wKp[13] = 256; wKt[13] = 256;
  wSrc[14] = (const float*)d_in[57]; bSrc[14] = (const float*)d_in[58];
  wNp[14] = 48;  wNt[14] = 40;  wKp[14] = 128; wKt[14] = 128;

  size_t o_W[15];
  for (int i = 0; i < 15; ++i) o_W[i] = alloc((size_t)wNp[i] * wKp[i] * 2);

  for (int i = 0; i < 15; ++i) {
    int tot = wNp[i] * wKp[i];
    k_convw<<<(tot + 255) / 256, 256, 0, stream>>>(wSrc[i], (u16*)(ws + o_W[i]),
                                                   wNp[i], wKp[i], wNt[i], wKt[i]);
  }
  k_padbias<<<1, 64, 0, stream>>>(bSrc[14], (float*)(ws + o_bias48), 40, 48);

  const float* xyz = (const float*)d_in[0];
  k_transpose<<<(B * 4096 + 255) / 256, 256, 0, stream>>>(xyz, (float*)(ws + o_pts0), 4096);

  auto run_gemm = [&](const u16* Xin, int li, u16* Y, int M) {
    int tilesM = M / 16, tilesN = wNp[li] / 16;
    dim3 g((tilesM + 3) / 4, tilesN), blk(128);
    const float* bias = (li == 14) ? (const float*)(ws + o_bias48) : bSrc[li];
    k_gemm_bf16<<<g, blk, 0, stream>>>(Xin, (const u16*)(ws + o_W[li]), bias, Y,
                                       M, wKp[li], wNp[li]);
  };
  auto run_bn = [&](u16* Y, int M, int C, const float* gamma, const float* beta) {
    float* stats = (float*)(ws + o_stats);
    k_zero_stats<<<(2 * C + 255) / 256, 256, 0, stream>>>(stats, C);
    int chunks = (M + 256 * 16 - 1) / (256 * 16);
    dim3 g(C, chunks);
    k_stats_acc<<<g, 256, 0, stream>>>(Y, stats, M, C);
    long tot = (long)M * C;
    k_bnrelu<<<(int)((tot + 255) / 256), 256, 0, stream>>>(Y, stats, gamma, beta, (long)M, C);
  };

  const float* ptsPrev = (const float*)(ws + o_pts0);
  const u16* featsPrev = nullptr;
  for (int s = 0; s < 4; ++s) {
    int N = Ns[s], S = Ss[s];
    int* fps = (int*)(ws + o_fps);
    k_fps<<<B, 256, 0, stream>>>(ptsPrev, N, S, fps);
    float* nxyz = (float*)(ws + o_nxyz[s]);
    k_gather_xyz<<<(B * S + 255) / 256, 256, 0, stream>>>(ptsPrev, fps, nxyz, N, S);
    int* idx = (int*)(ws + o_idx);
    k_ballquery<<<(B * S + 255) / 256, 256, 0, stream>>>(ptsPrev, nxyz, idx, N, S, R2[s]);

    int M = B * S * 32;
    u16* X = (u16*)(ws + o_X);
    long totX = (long)M * Cpad0[s];
    k_group<<<(int)((totX + 255) / 256), 256, 0, stream>>>(ptsPrev, nxyz, featsPrev, idx,
                                                           X, N, S, Cfeat[s], Cpad0[s], totX);
    u16* Y1 = (u16*)(ws + o_Y1);
    u16* Y2 = (u16*)(ws + o_Y2);
    int lb = s * 3, pb = 1 + s * 12;
    run_gemm(X, lb + 0, Y1, M);
    run_bn(Y1, M, CH[s][0], (const float*)d_in[pb + 0 * 4 + 2], (const float*)d_in[pb + 0 * 4 + 3]);
    run_gemm(Y1, lb + 1, Y2, M);
    run_bn(Y2, M, CH[s][1], (const float*)d_in[pb + 1 * 4 + 2], (const float*)d_in[pb + 1 * 4 + 3]);
    run_gemm(Y2, lb + 2, Y1, M);
    run_bn(Y1, M, CH[s][2], (const float*)d_in[pb + 2 * 4 + 2], (const float*)d_in[pb + 2 * 4 + 3]);

    u16* feats = (u16*)(ws + o_feats[s]);
    k_maxpool_k<<<(B * S * CH[s][2] + 255) / 256, 256, 0, stream>>>(Y1, feats, B * S, CH[s][2]);
    ptsPrev = nxyz;
    featsPrev = feats;
  }

  // ----- classification head -----
  u16* pooled = (u16*)(ws + o_pool);
  k_maxpool_s<<<(16 * 512 + 255) / 256, 256, 0, stream>>>(featsPrev, pooled, 16, 512);
  u16* h0 = (u16*)(ws + o_hY[0]);
  u16* h1 = (u16*)(ws + o_hY[1]);
  u16* h2 = (u16*)(ws + o_hY[2]);
  run_gemm(pooled, 12, h0, 16);
  run_bn(h0, 16, 256, (const float*)d_in[51], (const float*)d_in[52]);
  run_gemm(h0, 13, h1, 16);
  run_bn(h1, 16, 128, (const float*)d_in[55], (const float*)d_in[56]);
  run_gemm(h1, 14, h2, 16);
  k_final<<<(640 + 255) / 256, 256, 0, stream>>>(h2, (float*)d_out);
}
